// GNN_Encoder_8598524527201
// MI455X (gfx1250) — matrix operand order repaired and checked by reading, not compiled, not run on hardware
//
#include <hip/hip_runtime.h>
#include <cstddef>

// ---------------- problem constants (from reference) ----------------
#define N_CELL   60000
#define N_GENE   4000
#define DIM      128
#define N_EDGE   1500000
#define LAYERS   2
#define EPS_BN   1e-5f

typedef __attribute__((ext_vector_type(16))) __bf16 v16bf;
typedef __attribute__((ext_vector_type(8)))  __bf16 v8bf;
typedef __attribute__((ext_vector_type(8)))  float  v8f;

// round-to-nearest-even float -> bf16 bits
__device__ __forceinline__ unsigned short f2bf(float f) {
    union { float f; unsigned int u; } x; x.f = f;
    unsigned int u = x.u;
    unsigned int rnd = 0x7FFFu + ((u >> 16) & 1u);
    return (unsigned short)((u + rnd) >> 16);
}

// ---------------- k0: zero accumulators ----------------
__global__ void k0_zero(float* __restrict__ p, size_t n) {
    size_t i = (size_t)blockIdx.x * blockDim.x + threadIdx.x;
    if (i < n) p[i] = 0.0f;
}

// ---------------- k1: edge scatter (segment-sum + count) ----------------
// one wave32 per edge; each lane atomically adds 4 floats of the gene row
__global__ void k1_scatter(const float* __restrict__ x_gene,
                           const int*   __restrict__ src,
                           const int*   __restrict__ dst,
                           float* __restrict__ agg,
                           float* __restrict__ cnt) {
    unsigned t = blockIdx.x * blockDim.x + threadIdx.x;
    unsigned e = t >> 5;
    unsigned lane = t & 31u;
    if (e >= N_EDGE) return;
    int s = src[e];
    int d = dst[e];
    const float4 v = *reinterpret_cast<const float4*>(x_gene + (size_t)s * DIM + lane * 4);
    float* o = agg + (size_t)d * DIM + lane * 4;
    atomicAdd(o + 0, v.x);
    atomicAdd(o + 1, v.y);
    atomicAdd(o + 2, v.z);
    atomicAdd(o + 3, v.w);
    if (lane == 0) atomicAdd(cnt + d, 1.0f);
}

// ---------------- k2: weights -> bf16, transposed (column-major) ----------------
// WlBt[n*DIM + k] = bf16(Wl[k*DIM + n]); same for Wr
__global__ void k2_convert_w(const float* __restrict__ Wl,
                             const float* __restrict__ Wr,
                             unsigned short* __restrict__ WlBt,
                             unsigned short* __restrict__ WrBt) {
    int idx = blockIdx.x * blockDim.x + threadIdx.x;
    if (idx >= 2 * DIM * DIM) return;
    int which = idx / (DIM * DIM);
    int r = idx - which * DIM * DIM;
    int n = r >> 7, k = r & 127;
    const float* W = which ? Wr : Wl;
    unsigned short* O = which ? WrBt : WlBt;
    O[n * DIM + k] = f2bf(W[k * DIM + n]);
}

// ---------------- k3: agg/cnt -> bf16, x_cell -> bf16 ----------------
__global__ void k3_finalize(const float* __restrict__ x_cell,
                            const float* __restrict__ agg,
                            const float* __restrict__ cnt,
                            unsigned short* __restrict__ aggB,
                            unsigned short* __restrict__ xcB) {
    size_t idx = (size_t)blockIdx.x * blockDim.x + threadIdx.x;
    if (idx >= (size_t)N_CELL * DIM) return;
    int row = (int)(idx >> 7);
    float c = cnt[row];
    if (c < 1.0f) c = 1.0f;
    aggB[idx] = f2bf(agg[idx] / c);
    xcB[idx]  = f2bf(x_cell[idx]);
}

// ---------------- k4: WMMA GEMM  Y = aggB@Wl + xcB@Wr + bias ----------------
// block = 256 threads = 8 waves; each wave computes one 16x16 tile.
// blockIdx.x selects the 16-row stripe (m0), wave id selects the 16-col tile (n0).
__global__ __launch_bounds__(256)
void k4_gemm_wmma(const unsigned short* __restrict__ aggB,
                  const unsigned short* __restrict__ xcB,
                  const unsigned short* __restrict__ WlBt,
                  const unsigned short* __restrict__ WrBt,
                  const float* __restrict__ bias,
                  float* __restrict__ Y) {
    const int wave = threadIdx.x >> 5;     // 0..7 -> N-tile
    const int lane = threadIdx.x & 31;
    const int half = lane >> 4;            // 0 or 1
    const int l15  = lane & 15;
    const int m0 = blockIdx.x * 16;
    const int n0 = wave * 16;

    v8f acc = {};                          // f32 accumulator (8 VGPRs)

    // A fragment (16x32 bf16, ISA layout): lane row = l15;
    // halves hold K = half*8..+8 and K = 16+half*8..+8 within each 32-chunk.
    const int baseA = half * 8;
    const unsigned short* aRow = aggB + (size_t)(m0 + l15) * DIM;
    const unsigned short* xRow = xcB  + (size_t)(m0 + l15) * DIM;
    // B fragment (32x16 bf16): lane column = l15, K = half*8..+8 and
    // K = 16+half*8..+8 within each 32-chunk (weights stored transposed/column-major).
    const unsigned short* wlC = WlBt + (size_t)(n0 + l15) * DIM + half * 8;
    const unsigned short* wrC = WrBt + (size_t)(n0 + l15) * DIM + half * 8;

#pragma unroll
    for (int kt = 0; kt < 4; ++kt) {
        // ---- agg @ Wl contribution ----
        {
            v8bf lo = *reinterpret_cast<const v8bf*>(aRow + kt * 32 + baseA);
            v8bf hi = *reinterpret_cast<const v8bf*>(aRow + kt * 32 + baseA + 16);
            v16bf a;
#pragma unroll
            for (int i = 0; i < 8; ++i) { a[i] = lo[i]; a[i + 8] = hi[i]; }
            v8bf blo = *reinterpret_cast<const v8bf*>(wlC + kt * 32);
            v8bf bhi = *reinterpret_cast<const v8bf*>(wlC + kt * 32 + 16);
            v16bf b;
#pragma unroll
            for (int i = 0; i < 8; ++i) { b[i] = blo[i]; b[i + 8] = bhi[i]; }
            acc = __builtin_amdgcn_wmma_f32_16x16x32_bf16(
                      false, a, false, b, (short)0, acc, false, false);
        }
        // ---- x_cell @ Wr contribution ----
        {
            v8bf lo = *reinterpret_cast<const v8bf*>(xRow + kt * 32 + baseA);
            v8bf hi = *reinterpret_cast<const v8bf*>(xRow + kt * 32 + baseA + 16);
            v16bf a;
#pragma unroll
            for (int i = 0; i < 8; ++i) { a[i] = lo[i]; a[i + 8] = hi[i]; }
            v8bf blo = *reinterpret_cast<const v8bf*>(wrC + kt * 32);
            v8bf bhi = *reinterpret_cast<const v8bf*>(wrC + kt * 32 + 16);
            v16bf b;
#pragma unroll
            for (int i = 0; i < 8; ++i) { b[i] = blo[i]; b[i + 8] = bhi[i]; }
            acc = __builtin_amdgcn_wmma_f32_16x16x32_bf16(
                      false, a, false, b, (short)0, acc, false, false);
        }
    }

    const float bv = bias[n0 + l15];
    // C/D layout: VGPR r -> M = r + half*8 ; lane -> N = l15
#pragma unroll
    for (int r = 0; r < 8; ++r) {
        int row = m0 + r + half * 8;
        Y[(size_t)row * DIM + n0 + l15] = acc[r] + bv;
    }
}

// ---------------- k5: per-column sum / sumsq ----------------
__global__ void k5_col_stats(const float* __restrict__ Y,
                             float* __restrict__ colsum,
                             float* __restrict__ colsq) {
    int col = threadIdx.x & 127;
    int sub = threadIdx.x >> 7;                       // 0..1
    int rstart = blockIdx.x * 2 + sub;
    int rstep  = gridDim.x * 2;
    float s = 0.0f, q = 0.0f;
    for (int r = rstart; r < N_CELL; r += rstep) {
        float v = Y[(size_t)r * DIM + col];
        s += v;
        q += v * v;
    }
    atomicAdd(colsum + col, s);
    atomicAdd(colsq  + col, q);
}

// ---------------- k6: BatchNorm apply (in place on d_out) ----------------
__global__ void k6_bn_apply(float* __restrict__ Y,
                            const float* __restrict__ colsum,
                            const float* __restrict__ colsq) {
    size_t idx = (size_t)blockIdx.x * blockDim.x + threadIdx.x;
    if (idx >= (size_t)N_CELL * DIM) return;
    int col = (int)(idx & 127);
    const float invN = 1.0f / (float)N_CELL;
    float mu  = colsum[col] * invN;
    float var = colsq[col] * invN - mu * mu;
    Y[idx] = (Y[idx] - mu) * rsqrtf(var + EPS_BN);
}

// ---------------- host launcher ----------------
extern "C" void kernel_launch(void* const* d_in, const int* in_sizes, int n_in,
                              void* d_out, int out_size, void* d_ws, size_t ws_size,
                              hipStream_t stream) {
    (void)in_sizes; (void)n_in; (void)out_size; (void)ws_size;

    const float* x_cell = (const float*)d_in[0];
    const float* x_gene = (const float*)d_in[1];
    const float* Wl_gc  = (const float*)d_in[2];
    const float* bl_gc  = (const float*)d_in[3];
    const float* Wr_gc  = (const float*)d_in[4];
    const int*   gc_src = (const int*)d_in[8];
    const int*   gc_dst = (const int*)d_in[9];

    // only the last layer's cell branch is live
    const float* Wl = Wl_gc + (size_t)(LAYERS - 1) * DIM * DIM;
    const float* Wr = Wr_gc + (size_t)(LAYERS - 1) * DIM * DIM;
    const float* bl = bl_gc + (size_t)(LAYERS - 1) * DIM;

    // workspace carve-up (all 16B aligned)
    float* agg    = (float*)d_ws;                       // N_CELL*DIM
    float* cnt    = agg + (size_t)N_CELL * DIM;         // N_CELL
    float* colsum = cnt + N_CELL;                       // DIM
    float* colsq  = colsum + DIM;                       // DIM
    unsigned short* aggB = (unsigned short*)(colsq + DIM);      // N_CELL*DIM
    unsigned short* xcB  = aggB + (size_t)N_CELL * DIM;         // N_CELL*DIM
    unsigned short* WlBt = xcB  + (size_t)N_CELL * DIM;         // DIM*DIM
    unsigned short* WrBt = WlBt + DIM * DIM;                    // DIM*DIM

    float* Y = (float*)d_out;

    // k0: zero agg + cnt + column stats
    {
        size_t zn = (size_t)N_CELL * DIM + N_CELL + 2 * DIM;
        unsigned blocks = (unsigned)((zn + 255) / 256);
        k0_zero<<<blocks, 256, 0, stream>>>(agg, zn);
    }
    // k1: edge scatter (one wave per edge)
    {
        unsigned long long threads = (unsigned long long)N_EDGE * 32ull;
        unsigned blocks = (unsigned)((threads + 255) / 256);
        k1_scatter<<<blocks, 256, 0, stream>>>(x_gene, gc_src, gc_dst, agg, cnt);
    }
    // k2: weights -> bf16 transposed
    k2_convert_w<<<(2 * DIM * DIM + 255) / 256, 256, 0, stream>>>(Wl, Wr, WlBt, WrBt);
    // k3: mean + bf16 conversion of activations
    {
        size_t n = (size_t)N_CELL * DIM;
        k3_finalize<<<(unsigned)((n + 255) / 256), 256, 0, stream>>>(x_cell, agg, cnt, aggB, xcB);
    }
    // k4: fused WMMA GEMM + bias -> Y
    k4_gemm_wmma<<<N_CELL / 16, 256, 0, stream>>>(aggB, xcB, WlBt, WrBt, bl, Y);
    // k5: column statistics
    k5_col_stats<<<240, 256, 0, stream>>>(Y, colsum, colsq);
    // k6: normalize in place
    {
        size_t n = (size_t)N_CELL * DIM;
        k6_bn_apply<<<(unsigned)((n + 255) / 256), 256, 0, stream>>>(Y, colsum, colsq);
    }
}
